// MultiHeadAttentionLayer_63608465654090
// MI455X (gfx1250) — matrix operand order repaired and checked by reading, not compiled, not run on hardware
//
#include <hip/hip_runtime.h>

typedef __bf16 bf16;
typedef bf16  v8bf  __attribute__((ext_vector_type(8)));
typedef bf16  v16bf __attribute__((ext_vector_type(16)));
typedef float v8f   __attribute__((ext_vector_type(8)));

#if defined(__HIP_DEVICE_COMPILE__) && __has_builtin(__builtin_amdgcn_global_load_async_to_lds_b128)
#define HAVE_ASYNC 1
#else
#define HAVE_ASYNC 0
#endif

#if defined(__HIP_DEVICE_COMPILE__) && __has_builtin(__builtin_amdgcn_update_dpp)
#define HAVE_DPP 1
#else
#define HAVE_DPP 0
#endif

#if HAVE_ASYNC
typedef int v4i __attribute__((ext_vector_type(4)));
typedef __attribute__((address_space(1))) v4i as1_v4i;
typedef __attribute__((address_space(3))) v4i as3_v4i;
#define ASYNC_CP16(lp, gp) \
  __builtin_amdgcn_global_load_async_to_lds_b128((as1_v4i*)(void*)(gp), (as3_v4i*)(void*)(lp), 0, 0)
#if __has_builtin(__builtin_amdgcn_s_wait_asynccnt)
#define WAIT_ASYNC(n) __builtin_amdgcn_s_wait_asynccnt(n)
#else
#define WAIT_ASYNC(n) asm volatile("s_wait_asynccnt %0" ::"i"(n) : "memory")
#endif
#else
#define WAIT_ASYNC(n)
#endif

__device__ __forceinline__ v8f wmma_bf16(v16bf a, v16bf b, v8f c) {
  return __builtin_amdgcn_wmma_f32_16x16x32_bf16(false, a, false, b, (short)0, c, false, false);
}

// A-matrix fragment (16x32 bf16) from row-major LDS tile.
__device__ __forceinline__ v16bf ld_afrag(const bf16* p, int ld, int row0, int k0) {
  const int lane = threadIdx.x & 31, half = lane >> 4, r = lane & 15;
  const bf16* rp = p + (row0 + r) * ld + k0 + half * 8;
  v8bf lo = *(const v8bf*)rp;
  v8bf hi = *(const v8bf*)(rp + 16);
  return __builtin_shufflevector(lo, hi, 0,1,2,3,4,5,6,7,8,9,10,11,12,13,14,15);
}

// B-matrix fragment (32x16 bf16) from LDS tile stored transposed (N rows x K cols, row-major).
__device__ __forceinline__ v16bf ld_bfragT(const bf16* p, int ld, int n0, int k0) {
  const int lane = threadIdx.x & 31, half = lane >> 4, r = lane & 15;
  const bf16* rp = p + (n0 + r) * ld + k0 + half * 8;
  v8bf lo = *(const v8bf*)rp;
  v8bf hi = *(const v8bf*)(rp + 16);
  return __builtin_shufflevector(lo, hi, 0,1,2,3,4,5,6,7,8,9,10,11,12,13,14,15);
}

// 16-lane butterfly reductions via DPP16 (rows are exactly 16-lane groups on CDNA5)
__device__ __forceinline__ float red16_max(float v) {
#if HAVE_DPP
  int x;
  x = __builtin_amdgcn_update_dpp(0, __float_as_int(v), 0xB1, 0xF, 0xF, true);   // quad_perm xor1
  v = fmaxf(v, __int_as_float(x));
  x = __builtin_amdgcn_update_dpp(0, __float_as_int(v), 0x4E, 0xF, 0xF, true);   // quad_perm xor2
  v = fmaxf(v, __int_as_float(x));
  x = __builtin_amdgcn_update_dpp(0, __float_as_int(v), 0x141, 0xF, 0xF, true);  // row_half_mirror
  v = fmaxf(v, __int_as_float(x));
  x = __builtin_amdgcn_update_dpp(0, __float_as_int(v), 0x140, 0xF, 0xF, true);  // row_mirror
  v = fmaxf(v, __int_as_float(x));
#else
  v = fmaxf(v, __shfl_xor(v, 1, 32));
  v = fmaxf(v, __shfl_xor(v, 2, 32));
  v = fmaxf(v, __shfl_xor(v, 4, 32));
  v = fmaxf(v, __shfl_xor(v, 8, 32));
#endif
  return v;
}
__device__ __forceinline__ float red16_sum(float v) {
#if HAVE_DPP
  int x;
  x = __builtin_amdgcn_update_dpp(0, __float_as_int(v), 0xB1, 0xF, 0xF, true);
  v += __int_as_float(x);
  x = __builtin_amdgcn_update_dpp(0, __float_as_int(v), 0x4E, 0xF, 0xF, true);
  v += __int_as_float(x);
  x = __builtin_amdgcn_update_dpp(0, __float_as_int(v), 0x141, 0xF, 0xF, true);
  v += __int_as_float(x);
  x = __builtin_amdgcn_update_dpp(0, __float_as_int(v), 0x140, 0xF, 0xF, true);
  v += __int_as_float(x);
#else
  v += __shfl_xor(v, 1, 32);
  v += __shfl_xor(v, 2, 32);
  v += __shfl_xor(v, 4, 32);
  v += __shfl_xor(v, 8, 32);
#endif
  return v;
}

// ---------------- conversion / packing kernels ----------------
__global__ void k_cvt_f32_bf16(const float* __restrict__ in, bf16* __restrict__ out, int n) {
  int i = blockIdx.x * 256 + threadIdx.x;
  if (i < n) out[i] = (bf16)in[i];
}

// Wq/Wk/Wv [H=8, D=512, 64] -> transposed WqkvT [1536 rows(n) x 512 cols(d)], n = p*512+h*64+k
__global__ void k_pack_wqkvT(const float* __restrict__ Wq, const float* __restrict__ Wk,
                             const float* __restrict__ Wv, bf16* __restrict__ out) {
  int i = blockIdx.x * 256 + threadIdx.x;
  if (i >= 3 * 262144) return;
  int p = i >> 18, rem = i & 262143;
  int h = rem >> 15, d = (rem >> 6) & 511, k = rem & 63;
  const float* W = (p == 0) ? Wq : (p == 1) ? Wk : Wv;
  out[(size_t)(p * 512 + h * 64 + k) * 512 + d] = (bf16)W[rem];
}

// Wo [512 x 512] -> WoT [n x d] bf16
__global__ void k_transpose_wo(const float* __restrict__ in, bf16* __restrict__ out) {
  int i = blockIdx.x * 256 + threadIdx.x;
  if (i >= 262144) return;
  int d = i >> 9, n = i & 511;
  out[(size_t)n * 512 + d] = (bf16)in[i];
}

__global__ void k_pack_bias(const float* __restrict__ bq, const float* __restrict__ bk,
                            const float* __restrict__ bv, float* __restrict__ out) {
  int i = blockIdx.x * 256 + threadIdx.x;
  if (i >= 1536) return;
  int p = i / 512, r = i % 512;
  const float* b = (p == 0) ? bq : (p == 1) ? bk : bv;
  out[i] = b[r];
}

// ---------------- bf16 WMMA GEMM: C[MxN] = A[MxK] * BT[NxK]^T + bias ----------------
// 128x128 block tile, 256 threads (8 waves: 4 row-groups x 2 col-groups),
// K staged in 32-chunks, double-buffered LDS with async global->LDS copies.
__device__ __forceinline__ void load_tiles(const bf16* __restrict__ A, int lda, int m0,
                                           const bf16* __restrict__ BT, int ldbt, int n0,
                                           int k0, bf16* Asb, bf16* Bsb, int tid) {
#pragma unroll
  for (int idx = tid; idx < 512; idx += 256) {
    int row = idx >> 2, c = (idx & 3) * 8;
#if HAVE_ASYNC
    ASYNC_CP16(&Asb[row * 40 + c], A + (size_t)(m0 + row) * lda + k0 + c);
    ASYNC_CP16(&Bsb[row * 40 + c], BT + (size_t)(n0 + row) * ldbt + k0 + c);
#else
    *(v8bf*)(&Asb[row * 40 + c]) = *(const v8bf*)(A + (size_t)(m0 + row) * lda + k0 + c);
    *(v8bf*)(&Bsb[row * 40 + c]) = *(const v8bf*)(BT + (size_t)(n0 + row) * ldbt + k0 + c);
#endif
  }
}

template <bool OUT_F32>
__global__ __launch_bounds__(256) void gemm_bf16(const bf16* __restrict__ A, int lda,
                                                 const bf16* __restrict__ BT, int ldbt,
                                                 const float* __restrict__ bias,
                                                 void* __restrict__ C, int ldc, int K) {
  __shared__ bf16 As[2 * 128 * 40];
  __shared__ bf16 Bs[2 * 128 * 40];
  const int tid = threadIdx.x;
  const int m0 = blockIdx.y * 128, n0 = blockIdx.x * 128;
  const int w = tid >> 5, wm = w & 3, wn = w >> 2;

  v8f acc[2][4];
#pragma unroll
  for (int i = 0; i < 2; ++i)
#pragma unroll
    for (int j = 0; j < 4; ++j)
#pragma unroll
      for (int r = 0; r < 8; ++r) acc[i][j][r] = 0.0f;

  const int nk = K / 32;
  load_tiles(A, lda, m0, BT, ldbt, n0, 0, &As[0], &Bs[0], tid);
  for (int kt = 0; kt < nk; ++kt) {
    if (kt + 1 < nk)
      load_tiles(A, lda, m0, BT, ldbt, n0, (kt + 1) * 32,
                 &As[((kt + 1) & 1) * 5120], &Bs[((kt + 1) & 1) * 5120], tid);
#if HAVE_ASYNC
    if (kt + 1 < nk) { WAIT_ASYNC(4); } else { WAIT_ASYNC(0); }
#endif
    __syncthreads();
    const bf16* Asb = &As[(kt & 1) * 5120];
    const bf16* Bsb = &Bs[(kt & 1) * 5120];
    v16bf a0 = ld_afrag(Asb, 40, wm * 32, 0);
    v16bf a1 = ld_afrag(Asb, 40, wm * 32 + 16, 0);
#pragma unroll
    for (int nt = 0; nt < 4; ++nt) {
      v16bf bfr = ld_bfragT(Bsb, 40, wn * 64 + nt * 16, 0);
      acc[0][nt] = wmma_bf16(a0, bfr, acc[0][nt]);
      acc[1][nt] = wmma_bf16(a1, bfr, acc[1][nt]);
    }
    __syncthreads();
  }

  const int lane = tid & 31, half = lane >> 4, ln = lane & 15;
#pragma unroll
  for (int mt = 0; mt < 2; ++mt)
#pragma unroll
    for (int nt = 0; nt < 4; ++nt)
#pragma unroll
      for (int r = 0; r < 8; ++r) {
        int grow = m0 + wm * 32 + mt * 16 + r + half * 8;
        int gcol = n0 + wn * 64 + nt * 16 + ln;
        float v = acc[mt][nt][r] + bias[gcol];
        if (OUT_F32) ((float*)C)[(size_t)grow * ldc + gcol] = v;
        else         ((bf16*)C)[(size_t)grow * ldc + gcol] = (bf16)v;
      }
}

// ---------------- flash attention ----------------
// grid = (S/64, B*H), 128 threads (4 waves, each owns 16 query rows)
__global__ __launch_bounds__(128) void flash_attn(const bf16* __restrict__ qkv,
                                                  bf16* __restrict__ aout) {
  __shared__ bf16 Qs[64 * 72];
  __shared__ bf16 Ks[64 * 72];
  __shared__ bf16 Vs[64 * 72];   // transposed: Vs[vd*72 + t]
  __shared__ bf16 Ps[4 * 16 * 72];
  const int tid = threadIdx.x, lane = tid & 31, w = tid >> 5;
  const int half = lane >> 4, ln = lane & 15;
  const int bh = blockIdx.y, b = bh >> 3, h = bh & 7;
  const int r0 = blockIdx.x * 64;
  const size_t rowbase = (size_t)(b * 4096 + r0);

  // load Q tile (64 x 64)
#pragma unroll
  for (int idx = tid; idx < 512; idx += 128) {
    int row = idx >> 3, c = (idx & 7) * 8;
#if HAVE_ASYNC
    ASYNC_CP16(&Qs[row * 72 + c], qkv + (rowbase + row) * 1536 + h * 64 + c);
#else
    *(v8bf*)(&Qs[row * 72 + c]) = *(const v8bf*)(qkv + (rowbase + row) * 1536 + h * 64 + c);
#endif
  }
  WAIT_ASYNC(0);
  __syncthreads();
  v16bf aQ0 = ld_afrag(Qs, 72, w * 16, 0);
  v16bf aQ1 = ld_afrag(Qs, 72, w * 16, 32);

  float m[8], l[8];
  v8f o[4];
#pragma unroll
  for (int r = 0; r < 8; ++r) { m[r] = -3.0e38f; l[r] = 0.0f; }
#pragma unroll
  for (int vt = 0; vt < 4; ++vt)
#pragma unroll
    for (int r = 0; r < 8; ++r) o[vt][r] = 0.0f;

  for (int t0 = 0; t0 < 4096; t0 += 64) {
    __syncthreads();
    // K tile (async copy overlaps with V transpose below)
#pragma unroll
    for (int idx = tid; idx < 512; idx += 128) {
      int row = idx >> 3, c = (idx & 7) * 8;
      const bf16* gp = qkv + (size_t)(b * 4096 + t0 + row) * 1536 + 512 + h * 64 + c;
#if HAVE_ASYNC
      ASYNC_CP16(&Ks[row * 72 + c], gp);
#else
      *(v8bf*)(&Ks[row * 72 + c]) = *(const v8bf*)gp;
#endif
    }
    // V tile transposed into Vs[vd][t]
    for (int idx = tid; idx < 4096; idx += 128) {
      int t = idx >> 6, vd = idx & 63;
      Vs[vd * 72 + t] = qkv[(size_t)(b * 4096 + t0 + t) * 1536 + 1024 + h * 64 + vd];
    }
    // prefetch next V tile rows (global_prefetch_b8)
    if (t0 + 64 < 4096)
      __builtin_prefetch(qkv + (size_t)(b * 4096 + t0 + 64 + (tid >> 1)) * 1536 + 1024 + h * 64, 0, 0);
    WAIT_ASYNC(0);
    __syncthreads();

    // scores S = (Q K^T) * 1/sqrt(64)
    v8f s[4];
#pragma unroll
    for (int nt = 0; nt < 4; ++nt) {
      v8f z;
#pragma unroll
      for (int r = 0; r < 8; ++r) z[r] = 0.0f;
      v16bf bK0 = ld_bfragT(Ks, 72, nt * 16, 0);
      v16bf bK1 = ld_bfragT(Ks, 72, nt * 16, 32);
      z = wmma_bf16(aQ0, bK0, z);
      z = wmma_bf16(aQ1, bK1, z);
      s[nt] = z * 0.125f;
    }

    // online softmax (rows live in 16-lane halves; DPP16 butterflies)
    float mnew[8], alpha[8];
#pragma unroll
    for (int r = 0; r < 8; ++r) {
      float v = fmaxf(fmaxf(s[0][r], s[1][r]), fmaxf(s[2][r], s[3][r]));
      v = red16_max(v);
      mnew[r] = fmaxf(m[r], v);
      alpha[r] = __expf(m[r] - mnew[r]);
      m[r] = mnew[r];
    }
#pragma unroll
    for (int r = 0; r < 8; ++r) {
      float acc = 0.0f;
#pragma unroll
      for (int nt = 0; nt < 4; ++nt) {
        float p = __expf(s[nt][r] - mnew[r]);
        s[nt][r] = p;
        acc += p;
      }
      acc = red16_sum(acc);
      l[r] = l[r] * alpha[r] + acc;
    }
#pragma unroll
    for (int vt = 0; vt < 4; ++vt)
#pragma unroll
      for (int r = 0; r < 8; ++r) o[vt][r] *= alpha[r];

    // C-layout -> A-layout via per-wave LDS bounce (bf16 P)
    bf16* pw = &Ps[w * 16 * 72];
#pragma unroll
    for (int nt = 0; nt < 4; ++nt)
#pragma unroll
      for (int r = 0; r < 8; ++r)
        pw[(r + half * 8) * 72 + nt * 16 + ln] = (bf16)s[nt][r];
    asm volatile("s_wait_dscnt 0" ::: "memory");
    v16bf pA0 = ld_afrag(pw, 72, 0, 0);
    v16bf pA1 = ld_afrag(pw, 72, 0, 32);

    // O += P V
#pragma unroll
    for (int vt = 0; vt < 4; ++vt) {
      v16bf bV0 = ld_bfragT(Vs, 72, vt * 16, 0);
      v16bf bV1 = ld_bfragT(Vs, 72, vt * 16, 32);
      o[vt] = wmma_bf16(pA0, bV0, o[vt]);
      o[vt] = wmma_bf16(pA1, bV1, o[vt]);
    }
  }

  // epilogue: O * (1/l) -> aout bf16 [8192 x 512], col = h*64 + vd (heads concatenated)
  float inv[8];
#pragma unroll
  for (int r = 0; r < 8; ++r) inv[r] = 1.0f / l[r];
#pragma unroll
  for (int vt = 0; vt < 4; ++vt)
#pragma unroll
    for (int r = 0; r < 8; ++r) {
      int grow = r0 + w * 16 + r + half * 8;
      int gcol = h * 64 + vt * 16 + ln;
      aout[(size_t)(b * 4096 + grow) * 512 + gcol] = (bf16)(o[vt][r] * inv[r]);
    }
}

// ---------------- host launcher ----------------
extern "C" void kernel_launch(void* const* d_in, const int* in_sizes, int n_in,
                              void* d_out, int out_size, void* d_ws, size_t ws_size,
                              hipStream_t stream) {
  const float* x  = (const float*)d_in[0];
  const float* Wq = (const float*)d_in[1];
  const float* bq = (const float*)d_in[2];
  const float* Wk = (const float*)d_in[3];
  const float* bk = (const float*)d_in[4];
  const float* Wv = (const float*)d_in[5];
  const float* bv = (const float*)d_in[6];
  const float* Wo = (const float*)d_in[7];
  const float* bo = (const float*)d_in[8];
  float* out = (float*)d_out;

  char* ws = (char*)d_ws;
  bf16*  xb     = (bf16*)(ws);                            // 8192*512  bf16 = 8 MB
  bf16*  wqkvT  = (bf16*)(ws + 8388608);                  // 1536*512  bf16 = 1.5 MB
  float* bqkv   = (float*)(ws + 9961472);                 // 1536 f32
  bf16*  woT    = (bf16*)(ws + 9967616);                  // 512*512 bf16 = 0.5 MB
  bf16*  qkvbuf = (bf16*)(ws + 10491904);                 // 8192*1536 bf16 = 24 MB
  bf16*  aoutb  = (bf16*)(ws + 35657728);                 // 8192*512 bf16 = 8 MB

  k_cvt_f32_bf16<<<16384, 256, 0, stream>>>(x, xb, 8192 * 512);
  k_pack_wqkvT<<<3072, 256, 0, stream>>>(Wq, Wk, Wv, wqkvT);
  k_pack_bias<<<6, 256, 0, stream>>>(bq, bk, bv, bqkv);
  k_transpose_wo<<<1024, 256, 0, stream>>>(Wo, woT);

  // QKV projection: [8192,512] x [512,1536] + bias -> bf16 (B given transposed)
  gemm_bf16<false><<<dim3(12, 64), 256, 0, stream>>>(xb, 512, wqkvT, 512, bqkv,
                                                     (void*)qkvbuf, 1536, 512);
  // fused attention
  flash_attn<<<dim3(64, 16), 128, 0, stream>>>(qkvbuf, aoutb);

  // output projection: [8192,512] x [512,512] + bo -> f32 d_out
  gemm_bf16<true><<<dim3(4, 64), 256, 0, stream>>>(aoutb, 512, woT, 512, bo,
                                                   (void*)out, 512, 512);
}
